// TFAttention_5841155522670
// MI455X (gfx1250) — matrix operand order repaired and checked by reading, not compiled, not run on hardware
//
#include <hip/hip_runtime.h>

// ---------------------------------------------------------------------------
// GPT-2 attention block for MI455X (gfx1250, wave32, WMMA + TDM).
// bf16 inputs / f32 accumulate via v_wmma_f32_16x16x32_bf16.
// Non-transposed LDS tiles streamed by the Tensor Data Mover with hardware
// LDS padding, DOUBLE-BUFFERED (issue tile i+1, s_wait_tensorcnt 1, compute
// tile i). Transposed tiles staged manually.
// ---------------------------------------------------------------------------

typedef __attribute__((ext_vector_type(16))) __bf16 v16bf;
typedef __attribute__((ext_vector_type(8)))  __bf16 v8bf;
typedef __attribute__((ext_vector_type(8)))  float  v8f;
typedef unsigned int u32x4 __attribute__((ext_vector_type(4)));
typedef int          i32x8 __attribute__((ext_vector_type(8)));
typedef int          i32x4 __attribute__((ext_vector_type(4)));

#define WMMA_BF16(A, B, C) \
  __builtin_amdgcn_wmma_f32_16x16x32_bf16(false, (A), false, (B), (short)0, (C), false, false)

#if defined(__has_builtin)
#if __has_builtin(__builtin_amdgcn_tensor_load_to_lds)
#define HAVE_TDM 1
#endif
#endif
#ifndef HAVE_TDM
#define HAVE_TDM 0
#endif

static constexpr int Bsz = 4, S = 2048, D = 1024, H = 16, DH = 64;
static constexpr int QKV_N = 3 * D;              // 3072
static constexpr int MROWS = Bsz * S;            // 8192

static __device__ inline v16bf combine16(v8bf lo, v8bf hi) {
  v16bf r;
#pragma unroll
  for (int i = 0; i < 8; ++i) { r[i] = lo[i]; r[i + 8] = hi[i]; }
  return r;
}
// A-style fragment: 16 contiguous elems
static __device__ inline v16bf load_fragA(const __bf16* p) {
  return combine16(*(const v8bf*)p, *(const v8bf*)(p + 8));
}
// B-style fragment: 16 contiguous elems
static __device__ inline v16bf load_fragB(const __bf16* p) {
  return combine16(*(const v8bf*)p, *(const v8bf*)(p + 8));
}

static __device__ inline void wait_tensorcnt0() {
#if defined(__has_builtin) && __has_builtin(__builtin_amdgcn_s_wait_tensorcnt)
  __builtin_amdgcn_s_wait_tensorcnt(0);
#else
  asm volatile("s_wait_tensorcnt 0" ::: "memory");
#endif
}
static __device__ inline void wait_tensorcnt1() {
#if defined(__has_builtin) && __has_builtin(__builtin_amdgcn_s_wait_tensorcnt)
  __builtin_amdgcn_s_wait_tensorcnt(1);
#else
  asm volatile("s_wait_tensorcnt 1" ::: "memory");
#endif
}

// ---------------------------------------------------------------------------
// TDM: 2D tile load Global -> LDS (bf16, data_size=2B), optional LDS padding.
//   tile_x elems per row, tile_y rows, row stride stride_x elems.
//   pad_iv: pad after 2<<pad_iv dwords; pad_amt: insert (pad_amt+1) dwords.
// D# bit packing per CDNA5 ISA.
// ---------------------------------------------------------------------------
#if HAVE_TDM
static __device__ inline void tdm_load_2d_bf16(unsigned lds_off, const void* gptr,
                                               int tile_x, int tile_y, int stride_x,
                                               int pad_en, int pad_iv, int pad_amt) {
  unsigned long long ga = (unsigned long long)(uintptr_t)gptr;
  u32x4 g0;
  g0[0] = 1u;                                           // count=1, user descriptor
  g0[1] = lds_off;                                      // lds_addr
  g0[2] = (unsigned)(ga & 0xffffffffu);                 // global_addr[31:0]
  g0[3] = (unsigned)((ga >> 32) & 0x01ffffffu) | (2u << 30);  // addr[56:32], type=2
  i32x8 g1;
  g1[0] = (1 << 16) | (pad_en << 20) | (pad_iv << 22) | (pad_amt << 25);  // data_size=2B
  g1[1] = (tile_x & 0xffff) << 16;                      // tensor_dim0[15:0] (== tile_x)
  g1[2] = ((tile_x >> 16) & 0xffff) | ((tile_y & 0xffff) << 16);  // dim0 hi | dim1 lo
  g1[3] = ((tile_y >> 16) & 0xffff) | ((tile_x & 0xffff) << 16);  // dim1 hi | tile_dim0
  g1[4] = (tile_y & 0xffff);                            // tile_dim1 (tile_dim2=0)
  g1[5] = stride_x;                                     // tensor_dim0_stride[31:0]
  g1[6] = 0;
  g1[7] = 0;
  i32x4 z4 = {0, 0, 0, 0};
#if __clang_major__ >= 23
  i32x8 z8 = {0, 0, 0, 0, 0, 0, 0, 0};
  __builtin_amdgcn_tensor_load_to_lds(g0, g1, z4, z4, z8, 0);
#else
  __builtin_amdgcn_tensor_load_to_lds(g0, g1, z4, z4, 0);
#endif
}
static __device__ inline unsigned lds_off_of(const void* p) {
  return (unsigned)(uintptr_t)p;  // flat LDS address truncates to LDS byte offset
}
#endif

// ---------------------------------------------------------------------------
// f32 -> bf16 conversion
// ---------------------------------------------------------------------------
__global__ void cvt_f32_bf16(const float* __restrict__ in, __bf16* __restrict__ out, int n) {
  int i = (blockIdx.x * blockDim.x + threadIdx.x) * 4;
  if (i < n) {
    float4 v = *(const float4*)(in + i);
    out[i + 0] = (__bf16)v.x;
    out[i + 1] = (__bf16)v.y;
    out[i + 2] = (__bf16)v.z;
    out[i + 3] = (__bf16)v.w;
  }
}

// ---------------------------------------------------------------------------
// Tiled WMMA GEMM: C[M,N] = A[M,K] * B[K,N] + bias[N]
// Block tile 128x128, 256 threads = 8 waves (4x2 wave grid), K step 32.
// A tiles streamed by TDM, double-buffered; B transposed in LDS manually.
// ---------------------------------------------------------------------------
template <bool OUT_BF16>
__global__ __launch_bounds__(256) void gemm_bf16_kernel(
    const __bf16* __restrict__ A, const __bf16* __restrict__ B,
    const float* __restrict__ bias, void* __restrict__ Cout,
    int M, int N, int K) {
  constexpr int AST = 40;  // LDS row stride (elems): 32 + 4dw pad
  constexpr int BST = 40;
  __shared__ alignas(16) __bf16 Asm[2][128 * AST];  // ping-pong TDM buffers
  __shared__ alignas(16) __bf16 Bsm[128 * BST];     // [col][k]

  const int tid = threadIdx.x;
  const int lane = tid & 31, wid = tid >> 5;
  const int wr = wid >> 1, wc = wid & 1;
  const int l16 = lane & 15, lh = lane >> 4;
  const int m0 = blockIdx.y * 128;
  const int n0 = blockIdx.x * 128;

  v8f acc[2][4];
#pragma unroll
  for (int mt = 0; mt < 2; ++mt)
#pragma unroll
    for (int nt = 0; nt < 4; ++nt) acc[mt][nt] = (v8f)(0.f);

  const int nk = K / 32;
#if HAVE_TDM
  if (wid == 0) {  // prologue: DMA A tile 0 (128 rows x 32 bf16, pad => stride 40)
    tdm_load_2d_bf16(lds_off_of(&Asm[0][0]), A + (size_t)m0 * K,
                     32, 128, K, /*pad_en=*/1, /*pad_iv=*/3, /*pad_amt=*/3);
  }
#endif
  for (int ik = 0; ik < nk; ++ik) {
    const int kb = ik * 32;
    __syncthreads();  // previous compute done: Bsm and Asm[(ik+1)&1] are free
#if HAVE_TDM
    if (wid == 0 && ik + 1 < nk) {  // DMA next A tile into the other buffer
      tdm_load_2d_bf16(lds_off_of(&Asm[(ik + 1) & 1][0]),
                       A + (size_t)m0 * K + kb + 32,
                       32, 128, K, 1, 3, 3);
    }
#else
    {  // stage A tile 128x32 (coalesced 32B per thread)
      int row = tid >> 1, kc = (tid & 1) * 16;
      const __bf16* src = A + (size_t)(m0 + row) * K + kb + kc;
      *(v8bf*)&Asm[ik & 1][row * AST + kc]     = *(const v8bf*)src;
      *(v8bf*)&Asm[ik & 1][row * AST + kc + 8] = *(const v8bf*)(src + 8);
    }
#endif
    {  // stage B tile 32x128 transposed into Bsm[col][k]
      int kr = tid >> 3, cc = (tid & 7) * 16;
      const __bf16* src = B + (size_t)(kb + kr) * N + n0 + cc;
      if (kb + 32 < K) __builtin_prefetch(src + (size_t)32 * N, 0, 1);
      __bf16 tmp[16];
      *(v8bf*)tmp       = *(const v8bf*)src;
      *(v8bf*)(tmp + 8) = *(const v8bf*)(src + 8);
#pragma unroll
      for (int i = 0; i < 16; ++i) Bsm[(cc + i) * BST + kr] = tmp[i];
    }
#if HAVE_TDM
    if (wid == 0) {  // in-order TDM: <=1 outstanding => tile ik resident
      if (ik + 1 < nk) wait_tensorcnt1(); else wait_tensorcnt0();
    }
#endif
    __syncthreads();

    const __bf16* Acur = &Asm[ik & 1][0];
    v16bf af[2], bf_[4];
#pragma unroll
    for (int mt = 0; mt < 2; ++mt)
      af[mt] = load_fragA(&Acur[(wr * 32 + mt * 16 + l16) * AST + lh * 16]);
#pragma unroll
    for (int nt = 0; nt < 4; ++nt)
      bf_[nt] = load_fragB(&Bsm[(wc * 64 + nt * 16 + l16) * BST + lh * 16]);
#pragma unroll
    for (int mt = 0; mt < 2; ++mt)
#pragma unroll
      for (int nt = 0; nt < 4; ++nt)
        acc[mt][nt] = WMMA_BF16(af[mt], bf_[nt], acc[mt][nt]);
  }

  // epilogue: C layout -> lane holds col (l16), rows lh*8 + j
#pragma unroll
  for (int nt = 0; nt < 4; ++nt) {
    int col = n0 + wc * 64 + nt * 16 + l16;
    float bv = bias[col];
#pragma unroll
    for (int mt = 0; mt < 2; ++mt) {
#pragma unroll
      for (int j = 0; j < 8; ++j) {
        int row = m0 + wr * 32 + mt * 16 + lh * 8 + j;
        float v = acc[mt][nt][j] + bv;
        if (OUT_BF16)
          ((__bf16*)Cout)[(size_t)row * N + col] = (__bf16)v;
        else
          ((float*)Cout)[(size_t)row * N + col] = v;
      }
    }
  }
}

// ---------------------------------------------------------------------------
// Causal flash attention. Grid: (S/64, B*H), block 128 (4 waves).
// Each wave: 16 query rows x DH=64. Key blocks of 32: K tile streamed by TDM
// (double-buffered, hw padding); V tile transposed manually.
// qkv: bf16 [B*S][3D], out: bf16 [B*S][D] (merged heads).
// ---------------------------------------------------------------------------
__global__ __launch_bounds__(128) void flash_attn_kernel(
    const __bf16* __restrict__ qkv, __bf16* __restrict__ out) {
  constexpr int KTS = 72;  // K tile row stride (elems): 64 + 4dw pad
  constexpr int VTS = 40;  // Vt row stride
  constexpr int PTS = 40;  // P staging row stride
  __shared__ alignas(16) __bf16 Kt[2][32 * KTS];   // ping-pong [key][dh]
  __shared__ alignas(16) __bf16 Vt[64 * VTS];      // [dh][key]  (transposed)
  __shared__ alignas(16) __bf16 Pb[4][16 * PTS];   // per-wave P staging

  const int tid = threadIdx.x, lane = tid & 31, w = tid >> 5;
  const int l16 = lane & 15, lh = lane >> 4;
  const int qb = blockIdx.x;          // query block (64 rows)
  const int bh = blockIdx.y;
  const int b = bh >> 4, h = bh & 15;

  const size_t rs = (size_t)QKV_N;    // row stride in qkv
  const __bf16* Qp = qkv + (size_t)b * S * rs + h * DH;
  const __bf16* Kp = Qp + D;
  const __bf16* Vp = Qp + 2 * D;

  const int q0 = qb * 64 + w * 16;    // wave's first query row

  // Q fragments (dh 0..31 and 32..63), pre-scaled by 1/sqrt(64) = 2^-3 (exact)
  v16bf qf[2];
  {
    const __bf16* qrow = Qp + (size_t)(q0 + l16) * rs + lh * 16;
#pragma unroll
    for (int f = 0; f < 2; ++f) {
      v16bf t = load_fragA(qrow + f * 32);
#pragma unroll
      for (int i = 0; i < 16; ++i) t[i] = (__bf16)((float)t[i] * 0.125f);
      qf[f] = t;
    }
  }

  v8f o[4];
#pragma unroll
  for (int d = 0; d < 4; ++d) o[d] = (v8f)(0.f);
  float mi[8], li[8];
#pragma unroll
  for (int j = 0; j < 8; ++j) { mi[j] = -3.0e38f; li[j] = 0.f; }

  const int nkb = qb * 2 + 2;  // causal: keys < qb*64+64
#if HAVE_TDM
  if (w == 0) {  // prologue: DMA K tile 0 (32 keys x 64 bf16, pad => stride 72)
    tdm_load_2d_bf16(lds_off_of(&Kt[0][0]), Kp,
                     64, 32, QKV_N, /*pad_en=*/1, /*pad_iv=*/4, /*pad_amt=*/3);
  }
#endif
  for (int kb = 0; kb < nkb; ++kb) {
    const int kbase = kb * 32;
    __syncthreads();  // previous compute done: Vt and Kt[(kb+1)&1] are free
#if HAVE_TDM
    if (w == 0 && kb + 1 < nkb) {  // DMA next K tile into the other buffer
      tdm_load_2d_bf16(lds_off_of(&Kt[(kb + 1) & 1][0]),
                       Kp + (size_t)(kbase + 32) * rs,
                       64, 32, QKV_N, 1, 4, 3);
    }
#else
    {  // stage K tile (32 keys x 64 dh), row-major
      int key = tid >> 2, dc = (tid & 3) * 16;
      const __bf16* src = Kp + (size_t)(kbase + key) * rs + dc;
      *(v8bf*)&Kt[kb & 1][key * KTS + dc]     = *(const v8bf*)src;
      *(v8bf*)&Kt[kb & 1][key * KTS + dc + 8] = *(const v8bf*)(src + 8);
    }
#endif
    {  // stage V tile transposed -> Vt[dh][key]
      int key = tid >> 2, dc = (tid & 3) * 16;
      const __bf16* src = Vp + (size_t)(kbase + key) * rs + dc;
      __bf16 tmp[16];
      *(v8bf*)tmp       = *(const v8bf*)src;
      *(v8bf*)(tmp + 8) = *(const v8bf*)(src + 8);
#pragma unroll
      for (int i = 0; i < 16; ++i) Vt[(dc + i) * VTS + key] = tmp[i];
    }
#if HAVE_TDM
    if (w == 0) {  // in-order TDM: <=1 outstanding => tile kb resident
      if (kb + 1 < nkb) wait_tensorcnt1(); else wait_tensorcnt0();
    }
#endif
    __syncthreads();

    const __bf16* Kcur = &Kt[kb & 1][0];
    // scores: S = Q(16x64) * K^T(64x32) as two 16x16 accum tiles
    v8f s[2];
#pragma unroll
    for (int t = 0; t < 2; ++t) {
      v8f z = (v8f)(0.f);
      v16bf b0 = load_fragB(&Kcur[(t * 16 + l16) * KTS + lh * 16]);       // dh 0..31
      v16bf b1 = load_fragB(&Kcur[(t * 16 + l16) * KTS + 32 + lh * 16]);  // dh 32..63
      z = WMMA_BF16(qf[0], b0, z);
      z = WMMA_BF16(qf[1], b1, z);
      s[t] = z;
    }

    // causal mask (only when this key block can cross the diagonal)
    if (kbase + 31 > q0) {
#pragma unroll
      for (int t = 0; t < 2; ++t)
#pragma unroll
        for (int j = 0; j < 8; ++j) {
          int keyi = kbase + t * 16 + l16;
          int qi = q0 + lh * 8 + j;
          if (keyi > qi) s[t][j] = -1.0e30f;
        }
    }

    // online softmax (row m in VGPR j, spread over a 16-lane half)
    float al[8];
#pragma unroll
    for (int j = 0; j < 8; ++j) {
      float m2 = fmaxf(s[0][j], s[1][j]);
#pragma unroll
      for (int off = 1; off < 16; off <<= 1) m2 = fmaxf(m2, __shfl_xor(m2, off, 32));
      float mc = fmaxf(mi[j], m2);
      al[j] = __expf(mi[j] - mc);
      mi[j] = mc;
      s[0][j] = __expf(s[0][j] - mc);
      s[1][j] = __expf(s[1][j] - mc);
      float rsum = s[0][j] + s[1][j];
#pragma unroll
      for (int off = 1; off < 16; off <<= 1) rsum += __shfl_xor(rsum, off, 32);
      li[j] = li[j] * al[j] + rsum;
    }
#pragma unroll
    for (int d = 0; d < 4; ++d)
#pragma unroll
      for (int j = 0; j < 8; ++j) o[d][j] *= al[j];

    // P (C-layout f32) -> bf16 A-fragment via per-wave LDS staging
#pragma unroll
    for (int t = 0; t < 2; ++t)
#pragma unroll
      for (int j = 0; j < 8; ++j)
        Pb[w][(lh * 8 + j) * PTS + t * 16 + l16] = (__bf16)s[t][j];
    asm volatile("s_wait_dscnt 0" ::: "memory");  // same-wave LDS RAW fence

    v16bf pf = load_fragA(&Pb[w][l16 * PTS + lh * 16]);  // 16x32 P fragment

    // O += P(16x32) * V(32x64)
#pragma unroll
    for (int d = 0; d < 4; ++d) {
      v16bf vf = load_fragB(&Vt[(d * 16 + l16) * VTS + lh * 16]);
      o[d] = WMMA_BF16(pf, vf, o[d]);
    }
  }

  // epilogue: normalize and store merged-head bf16 output
#pragma unroll
  for (int j = 0; j < 8; ++j) {
    float inv = 1.f / li[j];
    int q = q0 + lh * 8 + j;
    size_t base = ((size_t)b * S + q) * (size_t)D + h * DH;
#pragma unroll
    for (int d = 0; d < 4; ++d)
      out[base + d * 16 + l16] = (__bf16)(o[d][j] * inv);
  }
}

// ---------------------------------------------------------------------------
// Orchestration
// ---------------------------------------------------------------------------
extern "C" void kernel_launch(void* const* d_in, const int* in_sizes, int n_in,
                              void* d_out, int out_size, void* d_ws, size_t ws_size,
                              hipStream_t stream) {
  const float* x      = (const float*)d_in[0];  // [B,S,D]
  const float* w_attn = (const float*)d_in[1];  // [D,3D]
  const float* b_attn = (const float*)d_in[2];  // [3D]
  const float* w_proj = (const float*)d_in[3];  // [D,D]
  const float* b_proj = (const float*)d_in[4];  // [D]
  float* outp = (float*)d_out;

  // workspace layout (bf16): x | w_attn | w_proj | qkv | attn_out  (~92 MB)
  __bf16* xb   = (__bf16*)d_ws;
  __bf16* wab  = xb   + (size_t)MROWS * D;
  __bf16* wpb  = wab  + (size_t)D * QKV_N;
  __bf16* qkvb = wpb  + (size_t)D * D;
  __bf16* ab   = qkvb + (size_t)MROWS * QKV_N;

  const int nx = MROWS * D, nwa = D * QKV_N, nwp = D * D;
  cvt_f32_bf16<<<dim3((nx / 4 + 255) / 256), dim3(256), 0, stream>>>(x, xb, nx);
  cvt_f32_bf16<<<dim3((nwa / 4 + 255) / 256), dim3(256), 0, stream>>>(w_attn, wab, nwa);
  cvt_f32_bf16<<<dim3((nwp / 4 + 255) / 256), dim3(256), 0, stream>>>(w_proj, wpb, nwp);

  // QKV projection: [8192,1024] x [1024,3072] -> bf16 qkv (+bias)
  gemm_bf16_kernel<true><<<dim3(QKV_N / 128, MROWS / 128), dim3(256), 0, stream>>>(
      xb, wab, b_attn, (void*)qkvb, MROWS, QKV_N, D);

  // causal attention, merged-head bf16 output
  flash_attn_kernel<<<dim3(S / 64, Bsz * H), dim3(128), 0, stream>>>(qkvb, ab);

  // output projection: [8192,1024] x [1024,1024] -> f32 d_out (+bias)
  gemm_bf16_kernel<false><<<dim3(D / 128, MROWS / 128), dim3(256), 0, stream>>>(
      ab, wpb, b_proj, (void*)outp, MROWS, D, D);
}
